// PV_RCNN_84499186581840
// MI455X (gfx1250) — matrix operand order repaired and checked by reading, not compiled, not run on hardware
//
#include <hip/hip_runtime.h>
#include <hip/hip_bf16.h>

typedef __attribute__((ext_vector_type(16))) _Float16 v16h;
typedef __attribute__((ext_vector_type(8)))  float    v8f;

#define N_KP 2048
#define HH 200
#define WW 200

// ---------------------------------------------------------------------------
// WMMA fragment helpers (CDNA5 16-bit layouts, wave32)
// ---------------------------------------------------------------------------
// A 16x32 f16 fragment from row-major LDS tile: per lane m=lane&15,
// K = (v/4)*16 + (lane>=16)*8 + (v%4)*2 (+1). K-pairs are contiguous, so the
// compiler merges this into two ds_load_b128.
__device__ inline v16h ldsA_frag(const _Float16* a, int lda, int lane) {
  const int m  = lane & 15;
  const int hi = lane >> 4;
  v16h r;
#pragma unroll
  for (int v = 0; v < 8; ++v) {
    const int k = ((v >> 2) << 4) + (hi << 3) + ((v & 3) << 1);
    r[2 * v]     = a[m * lda + k];
    r[2 * v + 1] = a[m * lda + k + 1];
  }
  return r;
}

// B fragment from pre-swizzled (fragment-major) LDS: 16 contiguous f16 per
// lane -> two ds_load_b128.
__device__ inline v16h ld_frag16(const _Float16* p) {
  return *(const v16h*)p;
}

__device__ inline void wave_lds_fence() {
  asm volatile("s_wait_dscnt 0" ::: "memory");
}

// ---------------------------------------------------------------------------
// Furthest point sampling: single persistent workgroup, dist[] lives in L2
// ---------------------------------------------------------------------------
__global__ void __launch_bounds__(1024) fps_kernel(
    const float* __restrict__ pts, float* __restrict__ dist,
    int* __restrict__ kp_idx, float* __restrict__ kp_xyz, int n, int k)
{
  __shared__ float rd[1024];
  __shared__ int   ri[1024];
  __shared__ float lxyz[3];
  const int t = threadIdx.x;

  for (int j = t; j < n; j += 1024) dist[j] = 3.402823466e38f;
  if (t == 0) { lxyz[0] = pts[0]; lxyz[1] = pts[1]; lxyz[2] = pts[2]; }
  __syncthreads();

  int last = 0;
  for (int i = 0; i < k; ++i) {
    const float lx = lxyz[0], ly = lxyz[1], lz = lxyz[2];
    if (t == 0) {
      kp_idx[i] = last;
      kp_xyz[3 * i] = lx; kp_xyz[3 * i + 1] = ly; kp_xyz[3 * i + 2] = lz;
    }
    float bd = -1.0f; int bi = 0x7fffffff;
    for (int j = t; j < n; j += 1024) {
      const float dx = pts[3 * j] - lx, dy = pts[3 * j + 1] - ly, dz = pts[3 * j + 2] - lz;
      const float d  = dx * dx + dy * dy + dz * dz;
      const float nm = fminf(dist[j], d);
      dist[j] = nm;
      if (nm > bd) { bd = nm; bi = j; }     // per-thread first (lowest j) max
    }
    rd[t] = bd; ri[t] = bi;
    __syncthreads();
    for (int s = 512; s > 0; s >>= 1) {     // argmax; ties -> lowest index
      if (t < s) {
        const float o = rd[t + s]; const int oi = ri[t + s];
        if (o > rd[t] || (o == rd[t] && oi < ri[t])) { rd[t] = o; ri[t] = oi; }
      }
      __syncthreads();
    }
    last = ri[0];
    __syncthreads();
    if (t == 0) {
      lxyz[0] = pts[3 * last];
      lxyz[1] = pts[3 * last + 1];
      lxyz[2] = pts[3 * last + 2];
    }
    __syncthreads();
  }
}

// ---------------------------------------------------------------------------
// Fused ball-query + grouping + 2-layer WMMA MLP + register maxpool.
// One wave32 per keypoint; 2 waves per block. f16 data, f32 accumulate.
// ---------------------------------------------------------------------------
template <int CIN, int C1, int C2, int NS>
__global__ void __launch_bounds__(64) sa_kernel(
    const float* __restrict__ sxyz, const float* __restrict__ sfeat, int nsrc,
    float r2,
    const float* __restrict__ W1, const float* __restrict__ B1,
    const float* __restrict__ W2, const float* __restrict__ B2,
    const float* __restrict__ kp_xyz, float* __restrict__ outp)
{
  constexpr int CK    = CIN + 3;
  constexpr int KP1   = ((CK + 31) / 32) * 32;   // K-padded input width
  constexpr int KP2   = ((C1 + 31) / 32) * 32;   // K-padded hidden width
  constexpr int MT    = NS / 16;
  constexpr int N1T   = C1 / 16;
  constexpr int N2T   = C2 / 16;
  constexpr int KT1   = KP1 / 32;
  constexpr int KT2   = KP2 / 32;
  constexpr int WAVES = 2;

  // Weights stored fragment-major: tile (kt,nt) -> 32 lanes x 16 contiguous f16
  __shared__ __align__(32) _Float16 sW1f[KT1 * N1T * 512];
  __shared__ __align__(32) _Float16 sW2f[KT2 * N2T * 512];
  __shared__ float    sB1v[C1];
  __shared__ float    sB2v[C2];
  __shared__ __align__(32) _Float16 sA[WAVES][NS * KP1];
  __shared__ __align__(32) _Float16 sH[WAVES][NS * KP2];

  const int tid  = threadIdx.x;
  const int lane = tid & 31;
  const int w    = tid >> 5;

  // Stage weights (f32 -> f16) pre-swizzled into the B-fragment layout:
  // element e of lane L in tile (kt,nt) is W[kt*32 + (e/8)*16 + (L>=16)*8 + (e%8)][nt*16 + (L&15)]
  for (int i = tid; i < KT1 * N1T * 512; i += 64) {
    const int e = i & 15, ls = (i >> 4) & 31, tile = i >> 9;
    const int nt = tile % N1T, kt = tile / N1T;
    const int kk = kt * 32 + (e >> 3) * 16 + (ls >> 4) * 8 + (e & 7);
    const int nn = nt * 16 + (ls & 15);
    sW1f[i] = (kk < CK) ? (_Float16)W1[kk * C1 + nn] : (_Float16)0.0f;
  }
  for (int i = tid; i < KT2 * N2T * 512; i += 64) {
    const int e = i & 15, ls = (i >> 4) & 31, tile = i >> 9;
    const int nt = tile % N2T, kt = tile / N2T;
    const int kk = kt * 32 + (e >> 3) * 16 + (ls >> 4) * 8 + (e & 7);
    const int nn = nt * 16 + (ls & 15);
    sW2f[i] = (kk < C1) ? (_Float16)W2[kk * C2 + nn] : (_Float16)0.0f;
  }
  for (int i = tid; i < C1; i += 64) sB1v[i] = B1[i];
  for (int i = tid; i < C2; i += 64) sB2v[i] = B2[i];
  // Zero group tiles once; K-padding columns are never overwritten afterwards.
  for (int i = tid; i < WAVES * NS * KP1; i += 64) (&sA[0][0])[i] = (_Float16)0.0f;
  for (int i = tid; i < WAVES * NS * KP2; i += 64) (&sH[0][0])[i] = (_Float16)0.0f;
  __syncthreads();

  const int gwave  = blockIdx.x * WAVES + w;
  const int nwaves = gridDim.x * WAVES;
  const int n0 = lane & 15;
  const int mb = (lane >> 4) * 8;

  for (int kpi = gwave; kpi < N_KP; kpi += nwaves) {
    const float kx = kp_xyz[3 * kpi];
    const float ky = kp_xyz[3 * kpi + 1];
    const float kz = kp_xyz[3 * kpi + 2];
    _Float16* A = &sA[w][0];

    // ---- ball query with wave32 ballot compaction ----
    int cnt = 0;
    float bestd = 3.402823466e38f; int bestj = 0;
    for (int base = 0; base < nsrc; base += 32) {
      if (cnt >= NS) break;
      const int j = base + lane;                 // nsrc is a multiple of 32
      const float dx = sxyz[3 * j] - kx;
      const float dy = sxyz[3 * j + 1] - ky;
      const float dz = sxyz[3 * j + 2] - kz;
      const float d2 = dx * dx + dy * dy + dz * dz;
      if (d2 < bestd) { bestd = d2; bestj = j; }
      const bool in = d2 <= r2;
      const unsigned mask = (unsigned)__ballot(in);
      const int row = cnt + __popc(mask & ((1u << lane) - 1u));
      if (in && row < NS) {
        _Float16* ar = A + row * KP1;
        ar[0] = (_Float16)dx; ar[1] = (_Float16)dy; ar[2] = (_Float16)dz;
        if constexpr (CIN % 4 == 0) {
#pragma unroll
          for (int c = 0; c < CIN; c += 4) {
            const float4 f = *(const float4*)(sfeat + j * CIN + c);
            ar[3 + c] = (_Float16)f.x; ar[4 + c] = (_Float16)f.y;
            ar[5 + c] = (_Float16)f.z; ar[6 + c] = (_Float16)f.w;
          }
        } else {
#pragma unroll
          for (int c = 0; c < CIN; ++c) ar[3 + c] = (_Float16)sfeat[j * CIN + c];
        }
      }
      cnt += __popc(mask);
    }
    // pad short groups with the nearest source point (ref: idx[:, :1])
    if (cnt < NS) {
#pragma unroll
      for (int off = 16; off > 0; off >>= 1) {
        const float od = __shfl_xor(bestd, off, 32);
        const int   oj = __shfl_xor(bestj, off, 32);
        if (od < bestd || (od == bestd && oj < bestj)) { bestd = od; bestj = oj; }
      }
      const int r = cnt + lane;
      if (r < NS) {
        _Float16* ar = A + r * KP1;
        ar[0] = (_Float16)(sxyz[3 * bestj] - kx);
        ar[1] = (_Float16)(sxyz[3 * bestj + 1] - ky);
        ar[2] = (_Float16)(sxyz[3 * bestj + 2] - kz);
        if constexpr (CIN % 4 == 0) {
#pragma unroll
          for (int c = 0; c < CIN; c += 4) {
            const float4 f = *(const float4*)(sfeat + bestj * CIN + c);
            ar[3 + c] = (_Float16)f.x; ar[4 + c] = (_Float16)f.y;
            ar[5 + c] = (_Float16)f.z; ar[6 + c] = (_Float16)f.w;
          }
        } else {
#pragma unroll
          for (int c = 0; c < CIN; ++c) ar[3 + c] = (_Float16)sfeat[bestj * CIN + c];
        }
      }
    }
    wave_lds_fence();

    // ---- layer 1: (NS x KP1) @ (KP1 x C1) + b, ReLU -> sH ----
#pragma unroll
    for (int mt = 0; mt < MT; ++mt) {
      v16h a[KT1];
#pragma unroll
      for (int kt = 0; kt < KT1; ++kt)
        a[kt] = ldsA_frag(A + mt * 16 * KP1 + kt * 32, KP1, lane);
#pragma unroll
      for (int nt = 0; nt < N1T; ++nt) {
        v8f c;
        const float bv = sB1v[nt * 16 + n0];
#pragma unroll
        for (int v = 0; v < 8; ++v) c[v] = bv;
#pragma unroll
        for (int kt = 0; kt < KT1; ++kt) {
          const v16h b = ld_frag16(&sW1f[((kt * N1T + nt) << 9) + (lane << 4)]);
          c = __builtin_amdgcn_wmma_f32_16x16x32_f16(false, a[kt], false, b,
                                                     (short)0, c, false, false);
        }
#pragma unroll
        for (int v = 0; v < 8; ++v)
          sH[w][(mt * 16 + mb + v) * KP2 + nt * 16 + n0] =
              (_Float16)fmaxf(c[v], 0.0f);
      }
    }
    wave_lds_fence();

    // ---- layer 2 + register maxpool (relu(max) == max(relu)) ----
    v16h a2[MT][KT2];
#pragma unroll
    for (int mt = 0; mt < MT; ++mt)
#pragma unroll
      for (int kt = 0; kt < KT2; ++kt)
        a2[mt][kt] = ldsA_frag(&sH[w][mt * 16 * KP2 + kt * 32], KP2, lane);
#pragma unroll
    for (int nt = 0; nt < N2T; ++nt) {
      const float bv = sB2v[nt * 16 + n0];
      float m = -3.402823466e38f;
#pragma unroll
      for (int mt = 0; mt < MT; ++mt) {
        v8f c;
#pragma unroll
        for (int v = 0; v < 8; ++v) c[v] = bv;
#pragma unroll
        for (int kt = 0; kt < KT2; ++kt) {
          const v16h b = ld_frag16(&sW2f[((kt * N2T + nt) << 9) + (lane << 4)]);
          c = __builtin_amdgcn_wmma_f32_16x16x32_f16(false, a2[mt][kt], false, b,
                                                     (short)0, c, false, false);
        }
#pragma unroll
        for (int v = 0; v < 8; ++v) m = fmaxf(m, c[v]);   // rows mb..mb+7
      }
      // combine the two half-lane row groups (same output column n0)
      m = fmaxf(m, __shfl_xor(m, 16, 32));
      m = fmaxf(m, 0.0f);                                  // ReLU after pool
      if (lane < 16) outp[(nt * 16 + n0) * N_KP + kpi] = m;
    }
  }
}

// ---------------------------------------------------------------------------
// BEV bilinear grid-sample (zeros padding, align_corners=False semantics)
// ---------------------------------------------------------------------------
__global__ void __launch_bounds__(256) bev_kernel(
    const float* __restrict__ vol, const float* __restrict__ kp_xyz,
    float* __restrict__ out)
{
  const int kk = blockIdx.x * 256 + threadIdx.x;
  const int c  = blockIdx.y;
  if (kk >= N_KP) return;
  float ix = kp_xyz[3 * kk] * 2.5f;                 // (x - 0) / 0.4
  float iy = (kp_xyz[3 * kk + 1] + 40.0f) * 2.5f;   // (y + 40) / 0.4
  ix = fminf(fmaxf(ix, 0.0f), (float)(HH - 1));
  iy = fminf(fmaxf(iy, 0.0f), (float)(WW - 1));
  const float gx = ix * (2.0f / 198.0f) - 1.0f;
  const float gy = iy * (2.0f / 198.0f) - 1.0f;
  const float x = ((gx + 1.0f) * (float)WW - 1.0f) * 0.5f;
  const float y = ((gy + 1.0f) * (float)HH - 1.0f) * 0.5f;
  const float x0f = floorf(x), y0f = floorf(y);
  const int   x0 = (int)x0f,  y0 = (int)y0f;
  const float wx1 = x - x0f, wy1 = y - y0f;
  const float wx0 = 1.0f - wx1, wy0 = 1.0f - wy1;
  const float* v = vol + (size_t)c * (HH * WW);
  auto g = [&](int yi, int xi) -> float {
    const bool ok = (xi >= 0) && (xi < WW) && (yi >= 0) && (yi < HH);
    const int yc = min(max(yi, 0), HH - 1);
    const int xc = min(max(xi, 0), WW - 1);
    return ok ? v[yc * WW + xc] : 0.0f;
  };
  out[c * N_KP + kk] = g(y0, x0)         * wy0 * wx0 +
                       g(y0, x0 + 1)     * wy0 * wx1 +
                       g(y0 + 1, x0)     * wy1 * wx0 +
                       g(y0 + 1, x0 + 1) * wy1 * wx1;
}

// ---------------------------------------------------------------------------
extern "C" void kernel_launch(void* const* d_in, const int* in_sizes, int n_in,
                              void* d_out, int out_size, void* d_ws, size_t ws_size,
                              hipStream_t stream) {
  const float* pts   = (const float*)d_in[0];   // 32768 x 3
  const float* pfeat = (const float*)d_in[1];   // 32768 x 1
  const float* vx1   = (const float*)d_in[2];   // 16384 x 3
  const float* f1    = (const float*)d_in[3];   // 16384 x 16
  const float* vx2   = (const float*)d_in[4];   //  8192 x 3
  const float* f2    = (const float*)d_in[5];   //  8192 x 32
  const float* vx3   = (const float*)d_in[6];   //  4096 x 3
  const float* f3    = (const float*)d_in[7];   //  4096 x 64
  const float* bev   = (const float*)d_in[8];   // 128 x 200 x 200
  // mlp_params leaves: [level][radius][(W1,b1),(W2,b2)] -> d_in[9 + l*8 + r*4 + p]
  auto P = [&](int lvl, int rad, int p) -> const float* {
    return (const float*)d_in[9 + lvl * 8 + rad * 4 + p];
  };

  float* ws     = (float*)d_ws;
  float* kp_xyz = ws;                         // 2048*3 floats
  float* dist   = ws + N_KP * 3;              // 32768 floats
  int*   kp_idx = (int*)(ws + N_KP * 3 + 32768);

  float* out = (float*)d_out;

  // 1) FPS (single persistent workgroup)
  fps_kernel<<<dim3(1), dim3(1024), 0, stream>>>(pts, dist, kp_idx, kp_xyz,
                                                 32768, N_KP);

  // 2) SA modules: one wave per keypoint (1024 blocks x 2 waves = 2048 waves)
  const dim3 sg(1024), sb(64);
  // level 0: c_in=1, mlp 16->16, radii 0.4 / 0.8
  sa_kernel<1, 16, 16, 16><<<sg, sb, 0, stream>>>(pts, pfeat, 32768, 0.16f,
      P(0,0,0), P(0,0,1), P(0,0,2), P(0,0,3), kp_xyz, out + (size_t)0 * N_KP);
  sa_kernel<1, 16, 16, 32><<<sg, sb, 0, stream>>>(pts, pfeat, 32768, 0.64f,
      P(0,1,0), P(0,1,1), P(0,1,2), P(0,1,3), kp_xyz, out + (size_t)16 * N_KP);
  // level 1: c_in=16, mlp 16->16, radii 0.4 / 0.8
  sa_kernel<16, 16, 16, 16><<<sg, sb, 0, stream>>>(vx1, f1, 16384, 0.16f,
      P(1,0,0), P(1,0,1), P(1,0,2), P(1,0,3), kp_xyz, out + (size_t)32 * N_KP);
  sa_kernel<16, 16, 16, 32><<<sg, sb, 0, stream>>>(vx1, f1, 16384, 0.64f,
      P(1,1,0), P(1,1,1), P(1,1,2), P(1,1,3), kp_xyz, out + (size_t)48 * N_KP);
  // level 2: c_in=32, mlp 32->32, radii 0.8 / 1.2
  sa_kernel<32, 32, 32, 16><<<sg, sb, 0, stream>>>(vx2, f2, 8192, 0.64f,
      P(2,0,0), P(2,0,1), P(2,0,2), P(2,0,3), kp_xyz, out + (size_t)64 * N_KP);
  sa_kernel<32, 32, 32, 32><<<sg, sb, 0, stream>>>(vx2, f2, 8192, 1.44f,
      P(2,1,0), P(2,1,1), P(2,1,2), P(2,1,3), kp_xyz, out + (size_t)96 * N_KP);
  // level 3: c_in=64, mlp 64->64, radii 1.2 / 2.4
  sa_kernel<64, 64, 64, 16><<<sg, sb, 0, stream>>>(vx3, f3, 4096, 1.44f,
      P(3,0,0), P(3,0,1), P(3,0,2), P(3,0,3), kp_xyz, out + (size_t)128 * N_KP);
  sa_kernel<64, 64, 64, 32><<<sg, sb, 0, stream>>>(vx3, f3, 4096, 5.76f,
      P(3,1,0), P(3,1,1), P(3,1,2), P(3,1,3), kp_xyz, out + (size_t)192 * N_KP);

  // 3) BEV bilinear sample -> channels 256..383
  bev_kernel<<<dim3(8, 128), dim3(256), 0, stream>>>(bev, kp_xyz,
                                                     out + (size_t)256 * N_KP);
}
